// TemporalMLPSCM_85100482003448
// MI455X (gfx1250) — matrix operand order repaired and checked by reading, not compiled, not run on hardware
//
#include <hip/hip_runtime.h>
#include <math.h>

// ---------------- problem constants (match reference) ----------------
#define B_      512
#define T_      500
#define C_      10
#define F_      10
#define L_      4
#define H_      64
#define PERIOD_ 12

// ---------------- kernel tiling ----------------
#define ROWS    16          // batch rows per workgroup (one WMMA M-tile)
#define BLK     128         // 4 wave32s: wave w owns output columns [16w,16w+16)
#define K0PAD   96          // layer-0 K = C+H = 74, padded to 3 K-blocks of 32
#define KPADM   128         // layers 1..3: K = 2H = 128 = 4 K-blocks
#define NKB0    3
#define NKBn    4
#define NFRAG   (NKB0 + 3*NKBn)   // 15 K-blocks of weight fragments total

// ---------------- LDS layout (dynamic shared, ~273 KB < 320 KB/WGP) ----------------
#define OFF_BFRAG   0
#define SZ_BFRAG    (NFRAG*4*32*8*4)            // 61440 B : pre-swizzled f16 B fragments
#define OFF_ATILE   (OFF_BFRAG + SZ_BFRAG)
#define SZ_ATILE    (ROWS*KPADM*2)              // 4096 B  : tanh(inp) tile, f16
#define OFF_STATE   (OFF_ATILE + SZ_ATILE)
#define SZ_STATE    (L_*ROWS*H_*4)              // 16384 B : mem (current states), f32
#define OFF_HIST    (OFF_STATE + SZ_STATE)
#define SZ_HIST     (L_*PERIOD_*ROWS*H_*4)      // 196608 B: 12-slot history ring, f32
#define OFF_BIAS    (OFF_HIST + SZ_HIST)
#define SZ_BIAS     (L_*H_*4)                   // 1024 B
#define OFF_FLAG    (OFF_BIAS + SZ_BIAS)
#define SZ_FLAG     64                          // row-bad bitmask word (+pad)
#define SMEM_TOTAL  (OFF_FLAG + SZ_FLAG)        // 279616 B

typedef __attribute__((ext_vector_type(16))) _Float16 v16h;
typedef __attribute__((ext_vector_type(8)))  float    v8f;
typedef __attribute__((ext_vector_type(4)))  int      v4i;

union FragV { v4i q[2]; v16h h; };

// One 16x16 output tile: accumulate over K-blocks with v_wmma_f32_16x16x32_f16.
// A layout (ISA 7.12.2, 16-bit A 16x32): lane<16 -> M=lane, halves [kb..kb+7] and
// [kb+16..kb+23]; lane>=16 -> same M, +8 K offset. Both runs are one ds_load_b128.
// B fragments pre-packed so each lane reads 8 consecutive dwords (2x ds_load_b128).
template<int KPAD, int NKB_T>
__device__ __forceinline__ v8f mm_tile(const _Float16* __restrict__ aTile,
                                       const int* __restrict__ bfrag,
                                       int fbase, int wave, int lane) {
  v8f acc = {};
  const int lo16 = lane & 15;
  const int aK   = (lane < 16) ? 0 : 8;
  #pragma unroll
  for (int j = 0; j < NKB_T; ++j) {
    FragV a, b;
    const _Float16* ap = aTile + lo16 * KPAD + j * 32 + aK;    // 16B-aligned
    a.q[0] = *(const v4i*)(ap);
    a.q[1] = *(const v4i*)(ap + 16);
    const int* bp = bfrag + ((fbase + j) * 4 + wave) * (32 * 8) + lane * 8;
    b.q[0] = *(const v4i*)(bp);
    b.q[1] = *(const v4i*)(bp + 4);
    acc = __builtin_amdgcn_wmma_f32_16x16x32_f16(false, a.h, false, b.h,
                                                 (short)0, acc, false, false);
  }
  return acc;
}

// Cooperative A-tile build with compile-time KPAD (constant-divisor addressing).
template<int KPAD>
__device__ __forceinline__ void build_a(int k, int t, int r0, int tid,
                                        const float* __restrict__ causes,
                                        const float* __restrict__ stS,
                                        _Float16* __restrict__ aTile) {
  for (int q = tid; q < ROWS * KPAD; q += BLK) {
    int m  = q / KPAD;
    int kk = q - m * KPAD;
    float v;
    if (KPAD == K0PAD) {                 // layer 0: [causes, mem0], zero-padded
      if (kk < C_)           v = causes[((size_t)(r0 + m) * T_ + t) * C_ + kk];
      else if (kk < C_ + H_) v = stS[m * H_ + (kk - C_)];
      else { aTile[q] = (_Float16)0.f; continue; }
    } else {                             // layers 1..3: [h_prev(new), mem_k(old)]
      if (kk < H_) v = stS[(k - 1) * ROWS * H_ + m * H_ + kk];
      else         v = stS[k * ROWS * H_ + m * H_ + (kk - H_)];
    }
    aTile[q] = (_Float16)tanhf(v);
  }
}

__global__ __launch_bounds__(BLK, 1)
void scm_kernel(const float* __restrict__ causes, const float* __restrict__ W0,
                const float* __restrict__ b0,     const float* __restrict__ Wk,
                const float* __restrict__ bk,     const float* __restrict__ nl,
                const float* __restrict__ ne,     const int* __restrict__ idxX,
                const int* __restrict__ idxY,     float* __restrict__ xout,
                float* __restrict__ yout) {
  extern __shared__ char smem[];
  int*      bfrag  = (int*)(smem + OFF_BFRAG);
  _Float16* aTile  = (_Float16*)(smem + OFF_ATILE);
  float*    stS    = (float*)(smem + OFF_STATE);
  float*    hiS    = (float*)(smem + OFF_HIST);
  float*    biS    = (float*)(smem + OFF_BIAS);
  int*      badW   = (int*)(smem + OFF_FLAG);    // single bitmask word, bit m = row m bad

  const int tid  = threadIdx.x;
  const int wave = tid >> 5;
  const int lane = tid & 31;
  const int lo16 = lane & 15;
  const int mhi  = (lane < 16) ? 0 : 8;        // D tile: VGPR i -> M = mhi + i
  const int r0   = blockIdx.x * ROWS;          // this WG's batch rows

  // ---- one-time init: biases ----
  for (int q = tid; q < L_ * H_; q += BLK)
    biS[q] = (q < H_) ? b0[q] : bk[q - H_];

  // ---- one-time init: weights -> f16 WMMA B-fragments in LDS ----
  // B[K][n] = W[n][K]; per-lane dword v holds K = kb(v), kb(v)+1 at column n.
  for (int q = tid; q < NFRAG * 4 * 32 * 8; q += BLK) {
    int v  = q & 7;
    int ln = (q >> 3) & 31;
    int nt = (q >> 8) & 3;
    int fj = q >> 10;
    int k, j;
    if (fj < NKB0) { k = 0; j = fj; }
    else           { k = 1 + (fj - NKB0) / NKBn; j = (fj - NKB0) % NKBn; }
    int kb = j * 32 + ((ln < 16) ? 0 : 8) + 2 * v + ((v < 4) ? 0 : 8);
    int n  = nt * 16 + (ln & 15);
    float w0v, w1v;
    if (k == 0) {
      w0v = (kb     < C_ + H_) ? W0[n * (C_ + H_) + kb]     : 0.f;
      w1v = (kb + 1 < C_ + H_) ? W0[n * (C_ + H_) + kb + 1] : 0.f;
    } else {
      const float* W = Wk + (size_t)(k - 1) * H_ * 2 * H_;
      w0v = W[n * (2 * H_) + kb];
      w1v = W[n * (2 * H_) + kb + 1];
    }
    union { _Float16 h[2]; int i32; } pk;
    pk.h[0] = (_Float16)w0v;
    pk.h[1] = (_Float16)w1v;
    bfrag[q] = pk.i32;
  }

  // ---- one-time init: zero state + history ring (contiguous in LDS) ----
  for (int q = tid; q < (SZ_STATE + SZ_HIST) / 4; q += BLK) stS[q] = 0.f;
  __syncthreads();

  // ---- hoist time-invariant output-gather slots into registers ----
  int om0 = -1, of0 = 0, oid0 = 0, om1 = -1, of1 = 0, oid1 = 0;
  if (tid < ROWS * F_)       { om0 = tid / F_;        of0 = tid - om0 * F_;
                               oid0 = idxX[of0] & 255; }
  if (tid + BLK < ROWS * F_) { om1 = (tid + BLK) / F_; of1 = (tid + BLK) - om1 * F_;
                               oid1 = idxX[of1] & 255; }
  const int idy = idxY[0] & 255;

  for (int t = 0; t < T_; ++t) {
    const int slot = t % PERIOD_;

    // Prefetch next step's noise: per (t,k) this WG's 16 rows are one contiguous
    // 4KB block; wave w covers layer w, lanes stride 128B -> full coverage.
    if (t + 1 < T_) {
      size_t boff = ((size_t)((t + 1) * L_ + wave) * B_ + r0) * H_;
      __builtin_prefetch((const char*)(nl + boff) + lane * 128, 0, 1);
      __builtin_prefetch((const char*)(ne + boff) + lane * 128, 0, 1);
    }

    for (int k = 0; k < L_; ++k) {
      // ---- phase 1: A-tile = tanh([left, mem_k]) in f16 (cooperative) ----
      if (tid == 0) badW[0] = 0;
      if (k == 0) build_a<K0PAD>(0, t, r0, tid, causes, stS, aTile);
      else        build_a<KPADM>(k, t, r0, tid, causes, stS, aTile);
      __syncthreads();

      // ---- phase 2: WMMA GEMM tile + elementwise pre-update ----
      v8f acc = (k == 0)
          ? mm_tile<K0PAD, NKB0>(aTile, bfrag, 0, wave, lane)
          : mm_tile<KPADM, NKBn>(aTile, bfrag, NKB0 + (k - 1) * NKBn, wave, lane);

      const int   ncol = wave * 16 + lo16;
      const float bias = biS[k * H_ + ncol];
      const float* nlp = nl + ((size_t)(t * L_ + k) * B_ + r0 + mhi) * H_ + ncol;
      const float* nep = ne + ((size_t)(t * L_ + k) * B_ + r0 + mhi) * H_ + ncol;
      float hv[8];
      int   badMask = 0;                         // bit m set if element non-finite
      #pragma unroll
      for (int i = 0; i < 8; ++i) {
        int m = mhi + i;
        float sOld = stS[k * ROWS * H_ + m * H_ + ncol];
        float past = hiS[((k * PERIOD_ + slot) * ROWS + m) * H_ + ncol];
        float h = acc[i] + bias + 0.01f * nlp[i * H_]
                + 0.8f * sOld + 0.6f * past + 0.1f * nep[i * H_];
        hv[i] = h;
        badMask |= (!(fabsf(h) <= 3.0e38f)) ? (1 << m) : 0;
      }
      if (badMask) atomicOr(badW, badMask);      // single guarded ds_or, ~never taken
      __syncthreads();

      // ---- phase 3: finalize (zero bad rows, clamp), commit state + history ----
      const int rowsBad = badW[0];
      #pragma unroll
      for (int i = 0; i < 8; ++i) {
        int m = mhi + i;
        float h = ((rowsBad >> m) & 1) ? 0.f : hv[i];
        h = fminf(fmaxf(h, -50.f), 50.f);
        stS[k * ROWS * H_ + m * H_ + ncol] = h;
        hiS[((k * PERIOD_ + slot) * ROWS + m) * H_ + ncol] = h;
      }
      __syncthreads();
    }

    // ---- outputs: X = out[:, idx_X], y = out[:, idx_y[0]] ----
    if (om0 >= 0)
      xout[((size_t)(r0 + om0) * T_ + t) * F_ + of0] =
          stS[(oid0 >> 6) * ROWS * H_ + om0 * H_ + (oid0 & 63)];
    if (om1 >= 0)
      xout[((size_t)(r0 + om1) * T_ + t) * F_ + of1] =
          stS[(oid1 >> 6) * ROWS * H_ + om1 * H_ + (oid1 & 63)];
    if (tid < ROWS)
      yout[(size_t)(r0 + tid) * T_ + t] =
          stS[(idy >> 6) * ROWS * H_ + tid * H_ + (idy & 63)];
    // next layer-0 build only reads stS; its barrier orders outputs vs. writes
  }
}

extern "C" void kernel_launch(void* const* d_in, const int* in_sizes, int n_in,
                              void* d_out, int out_size, void* d_ws, size_t ws_size,
                              hipStream_t stream) {
  (void)in_sizes; (void)n_in; (void)out_size; (void)d_ws; (void)ws_size;
  const float* causes = (const float*)d_in[0];
  const float* W0     = (const float*)d_in[1];
  const float* b0     = (const float*)d_in[2];
  const float* Wk     = (const float*)d_in[3];
  const float* bk     = (const float*)d_in[4];
  const float* nl     = (const float*)d_in[5];
  const float* ne     = (const float*)d_in[6];
  const int*   idxX   = (const int*)d_in[7];
  const int*   idxY   = (const int*)d_in[8];
  float* xout = (float*)d_out;
  float* yout = xout + (size_t)B_ * T_ * F_;

  (void)hipFuncSetAttribute((const void*)scm_kernel,
                            hipFuncAttributeMaxDynamicSharedMemorySize, SMEM_TOTAL);
  scm_kernel<<<dim3(B_ / ROWS), dim3(BLK), SMEM_TOTAL, stream>>>(
      causes, W0, b0, Wk, bk, nl, ne, idxX, idxY, xout, yout);
}
